// POCO_26456998543764
// MI455X (gfx1250) — matrix operand order repaired and checked by reading, not compiled, not run on hardware
//
#include <hip/hip_runtime.h>
#include <hip/hip_bf16.h>

// ---------------------------------------------------------------------------
// Perceiver-rotary forward for MI455X (gfx1250), wave32 + WMMA bf16.
//   B=16, NI=4096, NL=512, NO=1024, DIM=512, DH=64, CROSS_H=1, SELF_H=8
// GEMMs: v_wmma_f32_16x16x32_bf16, LayerNorm fused into A-tile staging,
// GEGLU fused into FFN-1 epilogue, LDS traffic b128 (B-tile stores b64).
// Attention: flash-style online softmax; K/V tiles staged with
// GLOBAL_LOAD_ASYNC_TO_LDS_B128 (ASYNCcnt) when the toolchain exposes it.
// Workspace requirement: ~173 MB.
// ---------------------------------------------------------------------------

typedef __bf16  v16bf __attribute__((ext_vector_type(16)));
typedef __bf16  v8bf  __attribute__((ext_vector_type(8)));
typedef float   v8f   __attribute__((ext_vector_type(8)));
typedef int     ai4   __attribute__((vector_size(16)));   // async-builtin arg type

#if defined(__has_builtin)
#if __has_builtin(__builtin_amdgcn_global_load_async_to_lds_b128) && \
    __has_builtin(__builtin_amdgcn_s_wait_asynccnt)
#define USE_ASYNC_LDS 1
#endif
#endif
#ifndef USE_ASYNC_LDS
#define USE_ASYNC_LDS 0
#endif

#define DEV __device__ __forceinline__

DEV unsigned short bfbits(float f) {
    unsigned u = __builtin_bit_cast(unsigned, f);
    u += 0x7fffu + ((u >> 16) & 1u);            // round-to-nearest-even
    return (unsigned short)(u >> 16);
}
DEV unsigned pack2(unsigned short a, unsigned short b) {
    return (unsigned)a | ((unsigned)b << 16);
}
DEV float gelu_tanh(float x) {
    const float c0 = 0.7978845608028654f;       // sqrt(2/pi)
    float u = c0 * (x + 0.044715f * x * x * x);
    return 0.5f * x * (1.0f + tanhf(u));
}
DEV v16bf frag_from_lds(const unsigned short* base0, const unsigned short* base1) {
    v8bf a = __builtin_bit_cast(v8bf, *(const uint4*)base0);
    v8bf b = __builtin_bit_cast(v8bf, *(const uint4*)base1);
    return __builtin_shufflevector(a, b, 0, 1, 2, 3, 4, 5, 6, 7,
                                   8, 9, 10, 11, 12, 13, 14, 15);
}

// ---------------------------------------------------------------------------
// Fused [LayerNorm] -> GEMM (bf16 WMMA) -> [bias] [+GEGLU] [+residual]
// C[M,N] = act( LN?(A)[M,K] @ W[K,N] ) ; tile 64x32 per 256-thread block.
// Requires M%64==0, N%32==0, K%32==0 (true for every call below).
// ---------------------------------------------------------------------------
template <bool LN, bool BIAS, bool RES, bool GEGLU>
__global__ __launch_bounds__(256)
void gemm_wmma(const float* __restrict__ A, int lda,
               const float* __restrict__ W, const float* __restrict__ W2, int ldw,
               const float* __restrict__ bias, const float* __restrict__ bias2,
               const float* __restrict__ res, float* __restrict__ out, int ldc,
               const float* __restrict__ mean, const float* __restrict__ rstd,
               const float* __restrict__ lng, const float* __restrict__ lnb,
               int M, int N, int K)
{
    __shared__ __align__(16) unsigned short sA [64 * 32];  // [m][k] row-major
    __shared__ __align__(16) unsigned short sBt [32 * 32]; // [n][k] transposed
    __shared__ __align__(16) unsigned short sBt2[32 * 32];

    const int tid  = threadIdx.x;
    const int lane = tid & 31;
    const int wave = tid >> 5;
    const int lo   = lane & 15;
    const int hi   = lane >> 4;
    const int wm   = (wave >> 1) << 4;          // wave M offset in tile (0..48)
    const int wn   = (wave & 1) << 4;           // wave N offset in tile (0 or 16)
    const int m0   = blockIdx.y * 64;
    const int n0   = blockIdx.x * 32;

    // staging assignments
    const int arow = tid >> 2;                  // A: 64 rows, 4 threads/row
    const int acb  = (tid & 3) * 8;             // 8 consecutive K each
    const int bn   = tid & 31;                  // B: one N column per thread
    const int bkb  = (tid >> 5) * 4;            // 4 consecutive K each

    v8f acc  = {};
    v8f acc2 = {};

    for (int k0 = 0; k0 < K; k0 += 32) {
        // ---- stage A tile (64x32): 2x b128 load, LN, pack, 1x ds b128 ----
        {
            const float* ap = &A[(size_t)(m0 + arow) * lda + k0 + acb];
            float4 x0 = *(const float4*)ap;
            float4 x1 = *(const float4*)(ap + 4);
            if (LN) {
                const float  mu = mean[m0 + arow], rs = rstd[m0 + arow];
                const float4 g0 = *(const float4*)&lng[k0 + acb];
                const float4 g1 = *(const float4*)&lng[k0 + acb + 4];
                const float4 b0 = *(const float4*)&lnb[k0 + acb];
                const float4 b1 = *(const float4*)&lnb[k0 + acb + 4];
                x0.x = (x0.x - mu) * rs * g0.x + b0.x;
                x0.y = (x0.y - mu) * rs * g0.y + b0.y;
                x0.z = (x0.z - mu) * rs * g0.z + b0.z;
                x0.w = (x0.w - mu) * rs * g0.w + b0.w;
                x1.x = (x1.x - mu) * rs * g1.x + b1.x;
                x1.y = (x1.y - mu) * rs * g1.y + b1.y;
                x1.z = (x1.z - mu) * rs * g1.z + b1.z;
                x1.w = (x1.w - mu) * rs * g1.w + b1.w;
            }
            uint4 u;
            u.x = pack2(bfbits(x0.x), bfbits(x0.y));
            u.y = pack2(bfbits(x0.z), bfbits(x0.w));
            u.z = pack2(bfbits(x1.x), bfbits(x1.y));
            u.w = pack2(bfbits(x1.z), bfbits(x1.w));
            *(uint4*)&sA[arow * 32 + acb] = u;
            if (k0 + 32 < K) __builtin_prefetch(ap + 32, 0, 1);  // next K tile
        }
        // ---- stage B tile(s) transposed: coalesced b32 loads, ds b64 ----
        {
            const float* wp = &W[(size_t)(k0 + bkb) * ldw + n0 + bn];
            uint2 u;
            u.x = pack2(bfbits(wp[0]), bfbits(wp[(size_t)ldw]));
            u.y = pack2(bfbits(wp[(size_t)2 * ldw]), bfbits(wp[(size_t)3 * ldw]));
            *(uint2*)&sBt[bn * 32 + bkb] = u;
            if (GEGLU) {
                const float* wp2 = &W2[(size_t)(k0 + bkb) * ldw + n0 + bn];
                uint2 u2;
                u2.x = pack2(bfbits(wp2[0]), bfbits(wp2[(size_t)ldw]));
                u2.y = pack2(bfbits(wp2[(size_t)2 * ldw]),
                             bfbits(wp2[(size_t)3 * ldw]));
                *(uint2*)&sBt2[bn * 32 + bkb] = u2;
            }
            if (k0 + 32 < K) __builtin_prefetch(wp + (size_t)32 * ldw, 0, 1);
        }
        __syncthreads();

        // ---- fragments: two ds_load_b128 each (gfx1250 wave32 layouts) ----
        // A lane(lo,hi): row wm+lo, K in [16h,16h+16) -- contiguous in sA
        const v16bf af = frag_from_lds(&sA[(wm + lo) * 32 + hi * 16],
                                       &sA[(wm + lo) * 32 + hi * 16 + 8]);
        // B lane(lo,hi): col wn+lo, K in [16h,16h+16) -- contiguous in sBt
        const v16bf bfr = frag_from_lds(&sBt[(wn + lo) * 32 + hi * 16],
                                        &sBt[(wn + lo) * 32 + hi * 16 + 8]);
        acc = __builtin_amdgcn_wmma_f32_16x16x32_bf16(
                  false, af, false, bfr, (short)0, acc, false, false);
        if (GEGLU) {
            const v16bf bg = frag_from_lds(&sBt2[(wn + lo) * 32 + hi * 16],
                                           &sBt2[(wn + lo) * 32 + hi * 16 + 8]);
            acc2 = __builtin_amdgcn_wmma_f32_16x16x32_bf16(
                       false, af, false, bg, (short)0, acc2, false, false);
        }
        __syncthreads();
    }

    // ---- epilogue: C VGPR r -> M = wm + 8*hi + r, N = wn + lo ----
    #pragma unroll
    for (int r = 0; r < 8; ++r) {
        const int row = m0 + wm + hi * 8 + r;
        const int col = n0 + wn + lo;
        float v = acc[r];
        if (BIAS) v += bias[col];
        if (GEGLU) {
            float g = acc2[r] + bias2[col];
            v = v * gelu_tanh(g);
        }
        if (RES) v += res[(size_t)row * ldc + col];
        out[(size_t)row * ldc + col] = v;
    }
}

// ---------------------------------------------------------------------------
// Per-row mean / rstd over K=512 (one wave per row, 8 rows per block).
// ---------------------------------------------------------------------------
__global__ __launch_bounds__(256)
void row_stats_k(const float* __restrict__ x, float* __restrict__ mean,
                 float* __restrict__ rstd)
{
    const int wave = threadIdx.x >> 5, lane = threadIdx.x & 31;
    const int row  = blockIdx.x * 8 + wave;
    const float4* xr = (const float4*)(x + (size_t)row * 512);
    float s = 0.f, s2 = 0.f;
    #pragma unroll
    for (int i = 0; i < 4; ++i) {
        float4 v = xr[i * 32 + lane];
        s  += v.x + v.y + v.z + v.w;
        s2 += v.x * v.x + v.y * v.y + v.z * v.z + v.w * v.w;
    }
    #pragma unroll
    for (int off = 16; off > 0; off >>= 1) {
        s  += __shfl_xor(s,  off, 32);
        s2 += __shfl_xor(s2, off, 32);
    }
    if (lane == 0) {
        float mu  = s * (1.f / 512.f);
        float var = s2 * (1.f / 512.f) - mu * mu;
        mean[row] = mu;
        rstd[row] = rsqrtf(var + 1e-5f);
    }
}

// ---------------------------------------------------------------------------
// Interleaved RoPE: out[2d]=x0*cos-x1*sin, out[2d+1]=x1*cos+x0*sin,
// theta_d = sign * ts[row] * INV_FREQ[d]; INV_FREQ[d]=0 for d>=16 (identity).
// ---------------------------------------------------------------------------
__global__ void rope_k(float* __restrict__ x, const float* __restrict__ ts,
                       int rows, int H, int rowStride, float sign)
{
    const int idx = blockIdx.x * blockDim.x + threadIdx.x;
    const int total = rows * H * 32;
    if (idx >= total) return;
    const int d = idx & 31;
    if (d >= 16) return;                         // zero-frequency half: identity
    const int t = idx >> 5;
    const int h = t % H, row = t / H;
    const float invf = 6.283185307179586f
                     / (0.1f * powf(1000.0f, (float)d * (1.0f / 16.0f)));
    const float th = sign * ts[row] * invf;
    const float c = cosf(th), s = sinf(th);
    float* p = x + (size_t)row * rowStride + h * 64 + 2 * d;
    const float x0 = p[0], x1 = p[1];
    p[0] = x0 * c - x1 * s;
    p[1] = x1 * c + x0 * s;
}

// ---------------------------------------------------------------------------
// Flash-style attention: one query row per thread, K/V tiles (64x64) in LDS,
// online softmax. grid = (NQ/256, H, B). mask (per b,key) may be null; a
// masked key gets score -1e30 (matches jax.where semantics incl. all-masked).
// out layout: (B, NQ, H, 64) contiguous. K/V staging uses the CDNA5 async
// global->LDS path (ASYNCcnt) when available.
// ---------------------------------------------------------------------------
__global__ __launch_bounds__(256)
void flash_attn_k(const float* __restrict__ q, int sq,
                  const float* __restrict__ k, int sk,
                  const float* __restrict__ v, int sv,
                  const unsigned char* __restrict__ mask,
                  float* __restrict__ out, int NQ, int NK, float scale)
{
    __shared__ __align__(16) float sK[64 * 64];
    __shared__ __align__(16) float sV[64 * 64];
    const int b = blockIdx.z, h = blockIdx.y, H = gridDim.y;
    const int tid = threadIdx.x;
    const int qi  = blockIdx.x * 256 + tid;

    float qreg[64], accv[64];
    const float* qrow = q + (size_t)(b * NQ + qi) * sq + h * 64;
    #pragma unroll
    for (int i = 0; i < 16; ++i) {
        float4 qv = *(const float4*)&qrow[4 * i];
        qreg[4 * i] = qv.x; qreg[4 * i + 1] = qv.y;
        qreg[4 * i + 2] = qv.z; qreg[4 * i + 3] = qv.w;
        accv[4 * i] = accv[4 * i + 1] = accv[4 * i + 2] = accv[4 * i + 3] = 0.f;
    }
    float m = -__builtin_inff(), l = 0.f;

    for (int j0 = 0; j0 < NK; j0 += 64) {
        __syncthreads();
        #pragma unroll
        for (int it = 0; it < 4; ++it) {
            const int vi  = it * 256 + tid;        // float4 index in 64x64 tile
            const int row = vi >> 4, c4 = (vi & 15) * 4;
            const float* ks = &k[(size_t)(b * NK + j0 + row) * sk + h * 64 + c4];
            const float* vs = &v[(size_t)(b * NK + j0 + row) * sv + h * 64 + c4];
#if USE_ASYNC_LDS
            __builtin_amdgcn_global_load_async_to_lds_b128(
                (ai4*)ks, (ai4*)&sK[row * 64 + c4], 0, 0);
            __builtin_amdgcn_global_load_async_to_lds_b128(
                (ai4*)vs, (ai4*)&sV[row * 64 + c4], 0, 0);
#else
            *(float4*)&sK[row * 64 + c4] = *(const float4*)ks;
            *(float4*)&sV[row * 64 + c4] = *(const float4*)vs;
#endif
        }
#if USE_ASYNC_LDS
        __builtin_amdgcn_s_wait_asynccnt(0);
#endif
        __syncthreads();
        for (int j = 0; j < 64; ++j) {
            float s = 0.f;
            #pragma unroll
            for (int i = 0; i < 16; ++i) {
                float4 kk = *(const float4*)&sK[j * 64 + 4 * i];
                s += qreg[4 * i] * kk.x + qreg[4 * i + 1] * kk.y
                   + qreg[4 * i + 2] * kk.z + qreg[4 * i + 3] * kk.w;
            }
            s *= scale;
            if (mask && !mask[(size_t)b * NK + j0 + j]) s = -1e30f;
            const float mn   = fmaxf(m, s);
            const float corr = __expf(m - mn);
            const float p    = __expf(s - mn);
            l = l * corr + p;
            #pragma unroll
            for (int i = 0; i < 16; ++i) {
                float4 vv = *(const float4*)&sV[j * 64 + 4 * i];
                accv[4 * i]     = accv[4 * i]     * corr + p * vv.x;
                accv[4 * i + 1] = accv[4 * i + 1] * corr + p * vv.y;
                accv[4 * i + 2] = accv[4 * i + 2] * corr + p * vv.z;
                accv[4 * i + 3] = accv[4 * i + 3] * corr + p * vv.w;
            }
            m = mn;
        }
    }
    const float inv = 1.f / l;
    float* orow = out + (size_t)((b * NQ + qi) * H + h) * 64;
    #pragma unroll
    for (int i = 0; i < 16; ++i) {
        float4 o;
        o.x = accv[4 * i] * inv;     o.y = accv[4 * i + 1] * inv;
        o.z = accv[4 * i + 2] * inv; o.w = accv[4 * i + 3] * inv;
        *(float4*)&orow[4 * i] = o;
    }
}

// ---------------------------------------------------------------------------
// Host orchestration
// ---------------------------------------------------------------------------
namespace {
constexpr int B = 16, NI = 4096, NL = 512, NO = 1024, DIM = 512;
constexpr int M_LAT = B * NL;      // 8192
constexpr int M_IN  = B * NI;      // 65536
constexpr int M_OQ  = B * NO;      // 16384
constexpr float SCALE = 0.125f;    // 64^-0.5

struct Ctx {
    hipStream_t s;
    float *lat, *mean, *rstd, *bufA, *bufQ, *bufO;
};

inline void stats(Ctx& c, const float* x, int rows) {
    row_stats_k<<<rows / 8, 256, 0, c.s>>>(x, c.mean, c.rstd);
}
inline void gemm_ln(Ctx& c, const float* A, const float* W, int ldw,
                    float* out, int ldc, const float* g, const float* b,
                    int M, int N) {
    dim3 gr(N / 32, M / 64);
    gemm_wmma<true, false, false, false><<<gr, 256, 0, c.s>>>(
        A, DIM, W, nullptr, ldw, nullptr, nullptr, nullptr, out, ldc,
        c.mean, c.rstd, g, b, M, N, DIM);
}
inline void gemm_bias_res(Ctx& c, const float* A, int lda, int K,
                          const float* W, const float* bias, float* io,
                          int M, int N) {
    dim3 gr(N / 32, M / 64);
    gemm_wmma<false, true, true, false><<<gr, 256, 0, c.s>>>(
        A, lda, W, nullptr, N, bias, nullptr, io, io, N,
        nullptr, nullptr, nullptr, nullptr, M, N, K);
}
inline void gemm_geglu(Ctx& c, const float* A, const float* w1,
                       const float* b1, float* out, const float* g,
                       const float* b, int M) {
    dim3 gr(2048 / 32, M / 64);
    gemm_wmma<true, true, false, true><<<gr, 256, 0, c.s>>>(
        A, DIM, w1, w1 + 2048, 4096, b1, b1 + 2048, nullptr, out, 2048,
        c.mean, c.rstd, g, b, M, 2048, DIM);
}
inline void rope(Ctx& c, float* x, const float* ts, int rows, int H,
                 int stride, float sign) {
    int total = rows * H * 32;
    rope_k<<<(total + 255) / 256, 256, 0, c.s>>>(x, ts, rows, H, stride, sign);
}
inline void ffn(Ctx& c, float* stream, int M, const float* const* P, int p0) {
    // P[p0..p0+5] = ln_g, ln_b, w1, b1, w2, b2
    stats(c, stream, M);
    gemm_geglu(c, stream, P[p0 + 2], P[p0 + 3], c.bufA, P[p0], P[p0 + 1], M);
    gemm_bias_res(c, c.bufA, 2048, 2048, P[p0 + 4], P[p0 + 5], stream, M, DIM);
}
} // namespace

extern "C" void kernel_launch(void* const* d_in, const int* in_sizes, int n_in,
                              void* d_out, int out_size, void* d_ws, size_t ws_size,
                              hipStream_t stream)
{
    // Params flattened in setup_inputs()/make_params() insertion order:
    //  0-7   enc_atn : ln_q_g ln_q_b ln_c_g ln_c_b wq wkv wo bo
    //  8-13  enc_ffn : ln_g ln_b w1 b1 w2 b2
    //  14-24 proc[0] : attn(ln_g ln_b wqkv wo bo) ffn(ln_g ln_b w1 b1 w2 b2)
    //  25-35 proc[1] : same
    //  36-43 dec_atn : ln_q_g ln_q_b ln_c_g ln_c_b wq wkv wo bo
    //  44-49 dec_ffn : ln_g ln_b w1 b1 w2 b2
    const float* P[50];
    for (int i = 0; i < 50; ++i) P[i] = (const float*)d_in[i];
    const float* inputs = (const float*)d_in[50];
    const float* lat_in = (const float*)d_in[51];
    const float* oq_in  = (const float*)d_in[52];
    const float* it_ts  = (const float*)d_in[53];
    const float* lt_ts  = (const float*)d_in[54];
    const float* ot_ts  = (const float*)d_in[55];
    const unsigned char* msk = (const unsigned char*)d_in[56];
    float* out = (float*)d_out;

    // Workspace layout (peak ~173 MB)
    char* w = (char*)d_ws;
    Ctx c;
    c.s    = stream;
    c.lat  = (float*)(w);                       // 16 MB  residual latent stream
    c.mean = (float*)(w + 16777216);            // 256 KB
    c.rstd = c.mean + 65536;                    // 256 KB
    c.bufA = (float*)(w + 17301504);            // 128 MB kv / qkv / geglu
    c.bufQ = (float*)(w + 151519232);           // 4 MB   cross-attn queries
    c.bufO = (float*)(w + 155713536);           // 16 MB  attention outputs

    (void)hipMemcpyAsync(c.lat, lat_in, (size_t)M_LAT * DIM * 4,
                         hipMemcpyDeviceToDevice, stream);
    (void)hipMemcpyAsync(out, oq_in, (size_t)M_OQ * DIM * 4,
                         hipMemcpyDeviceToDevice, stream);

    // ---------------- encoder cross-attn (H=1, rotate_value=True) ----------
    stats(c, c.lat, M_LAT);
    gemm_ln(c, c.lat, P[4], 64, c.bufQ, 64, P[0], P[1], M_LAT, 64);       // q
    stats(c, inputs, M_IN);
    gemm_ln(c, inputs, P[5], 128, c.bufA, 128, P[2], P[3], M_IN, 128);    // kv
    rope(c, c.bufQ,      lt_ts, M_LAT, 1, 64,  +1.f);                     // q
    rope(c, c.bufA,      it_ts, M_IN,  1, 128, +1.f);                     // k
    rope(c, c.bufA + 64, it_ts, M_IN,  1, 128, +1.f);                     // v
    flash_attn_k<<<dim3(NL / 256, 1, B), 256, 0, stream>>>(
        c.bufQ, 64, c.bufA, 128, c.bufA + 64, 128, msk, c.bufO, NL, NI, SCALE);
    rope(c, c.bufO, lt_ts, M_LAT, 1, 64, -1.f);                           // un-rope
    gemm_bias_res(c, c.bufO, 64, 64, P[6], P[7], c.lat, M_LAT, DIM);      // wo+res

    // ---------------- encoder FFN -----------------------------------------
    ffn(c, c.lat, M_LAT, P, 8);

    // ---------------- processor: 2x [self-attn(H=8) + FFN] ----------------
    const int bases[2] = {14, 25};
    for (int lyr = 0; lyr < 2; ++lyr) {
        const int a = bases[lyr];               // ln_g ln_b wqkv wo bo
        stats(c, c.lat, M_LAT);
        gemm_ln(c, c.lat, P[a + 2], 1536, c.bufA, 1536, P[a], P[a + 1],
                M_LAT, 1536);                                             // qkv
        rope(c, c.bufA,        lt_ts, M_LAT, 8, 1536, +1.f);              // q
        rope(c, c.bufA + 512,  lt_ts, M_LAT, 8, 1536, +1.f);              // k
        rope(c, c.bufA + 1024, lt_ts, M_LAT, 8, 1536, +1.f);              // v
        flash_attn_k<<<dim3(NL / 256, 8, B), 256, 0, stream>>>(
            c.bufA, 1536, c.bufA + 512, 1536, c.bufA + 1024, 1536,
            nullptr, c.bufO, NL, NL, SCALE);
        rope(c, c.bufO, lt_ts, M_LAT, 8, 512, -1.f);
        gemm_bias_res(c, c.bufO, 512, 512, P[a + 3], P[a + 4], c.lat,
                      M_LAT, DIM);
        ffn(c, c.lat, M_LAT, P, a + 5);
    }

    // ---------------- decoder cross-attn (H=1, rotate_value=False) --------
    stats(c, out, M_OQ);
    gemm_ln(c, out, P[40], 64, c.bufQ, 64, P[36], P[37], M_OQ, 64);       // q
    stats(c, c.lat, M_LAT);
    gemm_ln(c, c.lat, P[41], 128, c.bufA, 128, P[38], P[39], M_LAT, 128); // kv
    rope(c, c.bufQ, ot_ts, M_OQ, 1, 64,  +1.f);                           // q
    rope(c, c.bufA, lt_ts, M_LAT, 1, 128, +1.f);                          // k only
    flash_attn_k<<<dim3(NO / 256, 1, B), 256, 0, stream>>>(
        c.bufQ, 64, c.bufA, 128, c.bufA + 64, 128, nullptr, c.bufO,
        NO, NL, SCALE);
    gemm_bias_res(c, c.bufO, 64, 64, P[42], P[43], out, M_OQ, DIM);

    // ---------------- decoder FFN -----------------------------------------
    ffn(c, out, M_OQ, P, 44);

    (void)in_sizes; (void)n_in; (void)out_size; (void)ws_size;
}
